// GraphNetwork_19078244729183
// MI455X (gfx1250) — matrix operand order repaired and checked by reading, not compiled, not run on hardware
//
#include <hip/hip_runtime.h>

// ---------------------------------------------------------------------------
// GraphNetwork block for MI455X (gfx1250, wave32).
//  - GEMMs on v_wmma_f32_16x16x32_f16 (f16 A/B, f32 accumulate)
//  - Weights pre-converted + pre-swizzled ONCE into B-fragment byte order
//    in workspace (L2-resident), then streamed per-block into LDS with
//    global_load_async_to_lds_b128 (ASYNCcnt), double-buffered.
//  - segment_sums fused into producer kernels via global f32 atomics.
// ---------------------------------------------------------------------------

typedef _Float16 v16h __attribute__((ext_vector_type(16)));
typedef _Float16 h2   __attribute__((ext_vector_type(2)));
typedef float    v8f  __attribute__((ext_vector_type(8)));

#define GN_NODES   50000
#define GN_EDGES   800000
#define GN_GLOBALS 64
#define GN_HID     128
#define GN_LN_EPS  1e-5f

__device__ __forceinline__ v8f gn_zero8() { v8f z = {}; return z; }

// A-fragment (16x32 f16, MxK) from row-major f16 LDS buffer, ISA layout:
//   lane L: M = L&15, g = L>>4; VGPR v holds K = (v<4?0:16) + 8g + 2(v&3) + {0,1}
__device__ __forceinline__ v16h gn_load_a(const _Float16* buf, int strideH,
                                          int rowBase, int lane, int k0) {
  const int g = lane >> 4;
  const int row = rowBase + (lane & 15);
  const h2* p = (const h2*)buf;
  v16h a = {};
#pragma unroll
  for (int v = 0; v < 8; ++v) {
    const int koff = ((v < 4) ? 0 : 16) + 8 * g + 2 * (v & 3);
    h2 t = p[(row * strideH + k0 + koff) >> 1];
    a[2 * v]     = t.x;
    a[2 * v + 1] = t.y;
  }
  return a;
}

// Async copy one pre-packed 8KB weight chunk (global, contiguous) into LDS.
// 128 threads x 4 x b128 = 8192B; 4 async instructions outstanding per wave.
__device__ __forceinline__ void gn_async_fill(unsigned ldsDst,
                                              const _Float16* src, int tid) {
#pragma unroll
  for (int i = 0; i < 4; ++i) {
    unsigned d = ldsDst + (unsigned)(tid * 16 + i * 2048);
    const char* s = (const char*)src + tid * 16 + i * 2048;
    asm volatile("global_load_async_to_lds_b128 %0, %1, off"
                 :: "v"(d), "v"(s) : "memory");
  }
}

// One GEMM layer: acc[8] += A[64 x 32*KC] @ W[32*KC x 128], weight chunks
// double-buffered in LDS via async loads.
__device__ __forceinline__ void gn_gemm_layer(v8f acc[8], const _Float16* Wp,
                                              int KC, const _Float16* Abuf,
                                              int strideH, _Float16* Wl,
                                              unsigned wlOff, int tid,
                                              int lane, int wid) {
  gn_async_fill(wlOff, Wp, tid);                    // chunk 0 -> buf 0
  for (int kc = 0; kc < KC; ++kc) {
    __syncthreads();  // buf[(kc+1)&1] no longer read by any wave
    if (kc + 1 < KC) {
      gn_async_fill(wlOff + (unsigned)(((kc + 1) & 1) * 8192),
                    Wp + (size_t)(kc + 1) * 4096, tid);
      asm volatile("s_wait_asynccnt 4" ::: "memory");  // drain chunk kc only
    } else {
      asm volatile("s_wait_asynccnt 0" ::: "memory");
    }
    __syncthreads();  // chunk kc landed block-wide
    const _Float16* Wc = Wl + (kc & 1) * 4096;
    const v16h a = gn_load_a(Abuf, strideH, wid * 16, lane, kc * 32);
#pragma unroll
    for (int ct = 0; ct < 8; ++ct) {
      const v16h b = *((const v16h*)&Wc[ct * 512 + lane * 16]);
      acc[ct] = __builtin_amdgcn_wmma_f32_16x16x32_f16(
          false, a, false, b, (short)0, acc[ct], false, false);
    }
  }
}

// MODE 0: edges   in0=e, in1=x, in2=u           scatter: atom0[col], atom1[batch[row]]
// MODE 1: nodes   in0=x, in1=aggNode, in2=u     scatter: atom0[batch]
// MODE 2: globals in0=u, in1=nodeGlob, in2=edgeGlob
template <int DIN, int MODE>
__global__ __launch_bounds__(128) void gn_mlp_kernel(
    const float* __restrict__ in0, const float* __restrict__ in1,
    const float* __restrict__ in2,
    const int* __restrict__ idxRow, const int* __restrict__ idxCol,
    const int* __restrict__ batch,
    const _Float16* __restrict__ W1p, const float* __restrict__ b1g,
    const _Float16* __restrict__ W2p, const float* __restrict__ b2g,
    const float* __restrict__ gam, const float* __restrict__ bet,
    float* __restrict__ outp, float* __restrict__ atom0,
    float* __restrict__ atom1, int nRows) {
  constexpr int AS  = DIN + 8;        // padded f16 stride for input stage
  constexpr int HS  = GN_HID + 8;     // padded f16 stride for hidden stage
  constexpr int KC1 = DIN / 32;       // K-chunks layer 1 (10 or 9)

  __shared__ _Float16 AHs[64 * AS];   // input stage, reused as hidden stage
  __shared__ _Float16 Wl[2 * 4096];   // double-buffered 32x128 weight chunks
  __shared__ int sR[64], sC[64], sB[64];

  const int tid  = threadIdx.x;
  const int lane = tid & 31;
  const int wid  = tid >> 5;          // 4 waves; wave owns rows [16w, 16w+16)
  const int g    = lane >> 4;
  const int nsub = lane & 15;
  const int rowBlock = blockIdx.x * 64;
  const unsigned wlOff = (unsigned)(size_t)(&Wl[0]);  // LDS byte offset

  // ---- per-row gather indices -------------------------------------------
  if (tid < 64) {
    int gi = rowBlock + tid;
    gi = gi < nRows ? gi : (nRows - 1);
    if (MODE == 0) {
      const int s = idxRow[gi], c = idxCol[gi];
      sR[tid] = s; sC[tid] = c; sB[tid] = batch[s];
    } else if (MODE == 1) {
      sR[tid] = gi; sC[tid] = gi; sB[tid] = batch[gi];
    } else {
      sR[tid] = gi; sC[tid] = gi; sB[tid] = gi;
    }
  }
  __syncthreads();

  // ---- gather + f32->f16 convert input rows into LDS --------------------
  for (int idx = tid; idx < 64 * DIN; idx += 128) {
    const int r = idx / DIN;
    const int c = idx - r * DIN;
    int gi = rowBlock + r;
    gi = gi < nRows ? gi : (nRows - 1);
    float v;
    if (MODE == 0) {
      if (c < 32)       v = in0[(size_t)gi * 32 + c];
      else if (c < 160) v = in1[(size_t)sC[r] * 128 + (c - 32)];
      else if (c < 288) v = in1[(size_t)sR[r] * 128 + (c - 160)];
      else              v = in2[(size_t)sB[r] * 32 + (c - 288)];
    } else if (MODE == 1) {
      if (c < 128)      v = in0[(size_t)gi * 128 + c];
      else if (c < 256) v = in1[(size_t)gi * 128 + (c - 128)];
      else              v = in2[(size_t)sB[r] * 32 + (c - 256)];
    } else {
      if (c < 32)       v = in0[(size_t)gi * 32 + c];
      else if (c < 160) v = in1[(size_t)gi * 128 + (c - 32)];
      else              v = in2[(size_t)gi * 128 + (c - 160)];
    }
    AHs[r * AS + c] = (_Float16)v;
  }

  v8f acc[8];
#pragma unroll
  for (int ct = 0; ct < 8; ++ct) acc[ct] = gn_zero8();

  // ---- layer 1: [64 x DIN] @ [DIN x 128] --------------------------------
  gn_gemm_layer(acc, W1p, KC1, AHs, AS, Wl, wlOff, tid, lane, wid);

  // all A reads finished block-wide before reusing the buffer as H stage
  __syncthreads();

  // ---- bias1 + relu -> hidden stage (f16) -------------------------------
#pragma unroll
  for (int ct = 0; ct < 8; ++ct) {
    const float bias = b1g[ct * 16 + nsub];
#pragma unroll
    for (int r = 0; r < 8; ++r) {
      float h = acc[ct][r] + bias;
      h = h > 0.f ? h : 0.f;
      AHs[(wid * 16 + r + 8 * g) * HS + ct * 16 + nsub] = (_Float16)h;
    }
  }

#pragma unroll
  for (int ct = 0; ct < 8; ++ct) acc[ct] = gn_zero8();

  // ---- layer 2: [64 x 128] @ [128 x 128] --------------------------------
  gn_gemm_layer(acc, W2p, 4, AHs, HS, Wl, wlOff, tid, lane, wid);

  // ---- bias2 + relu + LayerNorm + store + fused segment-sum -------------
  float b2v[8], gv[8], bv[8];
#pragma unroll
  for (int ct = 0; ct < 8; ++ct) {
    const int n = ct * 16 + nsub;
    b2v[ct] = b2g[n]; gv[ct] = gam[n]; bv[ct] = bet[n];
  }
#pragma unroll
  for (int r = 0; r < 8; ++r) {
    // row M = r + 8g lives entirely in this 16-lane half-wave (8 cols/lane)
    float s = 0.f, ss = 0.f;
#pragma unroll
    for (int ct = 0; ct < 8; ++ct) {
      float h = acc[ct][r] + b2v[ct];
      h = h > 0.f ? h : 0.f;
      s += h; ss += h * h;
    }
#pragma unroll
    for (int off = 1; off < 16; off <<= 1) {   // stays inside the half-wave
      s  += __shfl_xor(s, off, 32);
      ss += __shfl_xor(ss, off, 32);
    }
    const float mean = s * (1.f / 128.f);
    const float var  = ss * (1.f / 128.f) - mean * mean;
    const float rstd = rsqrtf(var + GN_LN_EPS);
    const int lrow = wid * 16 + r + 8 * g;
    const int grow = rowBlock + lrow;
    const bool valid = grow < nRows;
#pragma unroll
    for (int ct = 0; ct < 8; ++ct) {
      float h = acc[ct][r] + b2v[ct];
      h = h > 0.f ? h : 0.f;
      const float y = gv[ct] * (h - mean) * rstd + bv[ct];
      const int n = ct * 16 + nsub;
      if (valid) {
        outp[(size_t)grow * 128 + n] = y;
        if (MODE == 0) {
          atomicAdd(&atom0[(size_t)sC[lrow] * 128 + n], y);  // edge->node
          atomicAdd(&atom1[(size_t)sB[lrow] * 128 + n], y);  // edge->global
        } else if (MODE == 1) {
          atomicAdd(&atom0[(size_t)sB[lrow] * 128 + n], y);  // node->global
        }
      }
    }
  }
}

// Pre-convert W[32*KC x 128] f32 -> f16 and swizzle into B-fragment byte
// order: chunk kc | frag ct | lane | 16 contiguous halves.
__global__ void gn_pack_w(const float* __restrict__ W,
                          _Float16* __restrict__ dst, int kchunks) {
  const int total = kchunks * 4096;
  for (int idx = blockIdx.x * blockDim.x + threadIdx.x; idx < total;
       idx += gridDim.x * blockDim.x) {
    const int kc   = idx >> 12;
    const int rem  = idx & 4095;
    const int ct   = rem >> 9;
    const int rem2 = rem & 511;
    const int ln   = rem2 >> 4;
    const int h    = rem2 & 15;
    const int k = kc * 32 + ((h >> 3) << 4) + ((ln >> 4) << 3) + (h & 7);
    const int n = (ct << 4) + (ln & 15);
    dst[idx] = (_Float16)W[(size_t)k * GN_HID + n];
  }
}

__global__ void gn_zero_kernel(float* p, size_t n) {
  size_t i = (size_t)blockIdx.x * blockDim.x + threadIdx.x;
  const size_t stride = (size_t)gridDim.x * blockDim.x;
  for (; i < n; i += stride) p[i] = 0.f;
}

extern "C" void kernel_launch(void* const* d_in, const int* in_sizes, int n_in,
                              void* d_out, int out_size, void* d_ws,
                              size_t ws_size, hipStream_t stream) {
  (void)in_sizes; (void)n_in; (void)out_size; (void)ws_size;
  const float* x     = (const float*)d_in[0];
  const float* e     = (const float*)d_in[1];
  const float* u     = (const float*)d_in[2];
  const int*   eidx  = (const int*)d_in[3];   // [2, N_EDGES]: row then col
  const int*   batch = (const int*)d_in[4];
  const float* eW1 = (const float*)d_in[5],  *eb1 = (const float*)d_in[6];
  const float* eW2 = (const float*)d_in[7],  *eb2 = (const float*)d_in[8];
  const float* eg  = (const float*)d_in[9],  *ebt = (const float*)d_in[10];
  const float* nW1 = (const float*)d_in[11], *nb1 = (const float*)d_in[12];
  const float* nW2 = (const float*)d_in[13], *nb2 = (const float*)d_in[14];
  const float* ng  = (const float*)d_in[15], *nbt = (const float*)d_in[16];
  const float* gW1 = (const float*)d_in[17], *gb1 = (const float*)d_in[18];
  const float* gW2 = (const float*)d_in[19], *gb2 = (const float*)d_in[20];
  const float* gg  = (const float*)d_in[21], *gbt = (const float*)d_in[22];

  float* out = (float*)d_out;
  float* x2 = out;                                       // [50000,128]
  float* e2 = out + (size_t)GN_NODES * 128;              // [800000,128]
  float* u2 = e2 + (size_t)GN_EDGES * 128;               // [64,128]

  // ---- workspace layout --------------------------------------------------
  float* aggNode = (float*)d_ws;                         // [50000,128]
  float* aggEG   = aggNode + (size_t)GN_NODES * 128;     // [64,128]
  float* aggNG   = aggEG + (size_t)GN_GLOBALS * 128;     // [64,128]
  _Float16* wp   = (_Float16*)(aggNG + (size_t)GN_GLOBALS * 128);
  _Float16* wpE1 = wp;                 // 10 chunks * 4096
  _Float16* wpE2 = wpE1 + 10 * 4096;   //  4 chunks
  _Float16* wpN1 = wpE2 + 4 * 4096;    //  9 chunks
  _Float16* wpN2 = wpN1 + 9 * 4096;    //  4 chunks
  _Float16* wpG1 = wpN2 + 4 * 4096;    //  9 chunks
  _Float16* wpG2 = wpG1 + 9 * 4096;    //  4 chunks

  const size_t zn = (size_t)GN_NODES * 128 + 2 * (size_t)GN_GLOBALS * 128;
  gn_zero_kernel<<<2048, 256, 0, stream>>>(aggNode, zn);

  // one-time weight convert + swizzle (L2-resident afterwards)
  gn_pack_w<<<40, 256, 0, stream>>>(eW1, wpE1, 10);
  gn_pack_w<<<16, 256, 0, stream>>>(eW2, wpE2, 4);
  gn_pack_w<<<36, 256, 0, stream>>>(nW1, wpN1, 9);
  gn_pack_w<<<16, 256, 0, stream>>>(nW2, wpN2, 4);
  gn_pack_w<<<36, 256, 0, stream>>>(gW1, wpG1, 9);
  gn_pack_w<<<16, 256, 0, stream>>>(gW2, wpG2, 4);

  // Edge MLP: 800000 rows, 64 rows/block
  gn_mlp_kernel<320, 0><<<GN_EDGES / 64, 128, 0, stream>>>(
      e, x, u, eidx, eidx + GN_EDGES, batch, wpE1, eb1, wpE2, eb2, eg, ebt, e2,
      aggNode, aggEG, GN_EDGES);

  // Node MLP
  gn_mlp_kernel<288, 1><<<(GN_NODES + 63) / 64, 128, 0, stream>>>(
      x, aggNode, u, nullptr, nullptr, batch, wpN1, nb1, wpN2, nb2, ng, nbt,
      x2, aggNG, nullptr, GN_NODES);

  // Global MLP: 64 rows in a single block
  gn_mlp_kernel<288, 2><<<1, 128, 0, stream>>>(
      u, aggNG, aggEG, nullptr, nullptr, nullptr, wpG1, gb1, wpG2, gb2, gg,
      gbt, u2, nullptr, nullptr, GN_GLOBALS);
}
